// CmntModel_24292335026241
// MI455X (gfx1250) — compile-verified, matrix operand order checked by reading, not run on hardware
//
#include <hip/hip_runtime.h>
#include <hip/hip_bf16.h>
#include <math.h>

// ---------------- model dims ----------------
#define DD   256
#define TD3  768        // 3*D
#define BB   32
#define TT   512
#define JJ   128
#define TWO_D 512

typedef __bf16 bf16_t;
typedef __attribute__((ext_vector_type(16))) __bf16 v16bf;
typedef __attribute__((ext_vector_type(2)))  __bf16 bf16x2;
typedef __attribute__((ext_vector_type(8)))  float  v8f;

// ---------------- helpers ----------------
__device__ __forceinline__ bf16_t f2bf(float f) {
    unsigned u = __float_as_uint(f);
    u += 0x7FFFu + ((u >> 16) & 1u);              // round-to-nearest-even
    unsigned short h = (unsigned short)(u >> 16);
    return __builtin_bit_cast(bf16_t, h);
}
__device__ __forceinline__ float sigm(float x) { return 1.0f / (1.0f + __expf(-x)); }

// Load a 16x32 bf16 WMMA operand fragment from a row-major source where the
// "lane index" dimension is rows (stride ld) and K runs along contiguous memory.
// Works for A (rows=M) and for B when the weight is stored [N][K] (rows=N),
// because (k,k+1) pairs are then adjacent -> single dword per element pair.
// Layout per ISA 7.12.2: lanes 0-15 -> K {0..7,16..23}, lanes 16-31 -> K {8..15,24..31}.
__device__ __forceinline__ v16bf load_frag16(const bf16_t* src, int ld) {
    const int lane = threadIdx.x & 31;
    const int m  = lane & 15;
    const int hf = lane >> 4;
    const unsigned int* p = reinterpret_cast<const unsigned int*>(src + (size_t)m * ld);
    v16bf f;
#pragma unroll
    for (int g = 0; g < 8; ++g) {
        const int kb = (g < 4) ? (2 * g + 8 * hf) : (16 + 2 * (g - 4) + 8 * hf);
        bf16x2 pr = __builtin_bit_cast(bf16x2, p[kb >> 1]);
        f[2 * g]     = pr[0];
        f[2 * g + 1] = pr[1];
    }
    return f;
}

__device__ __forceinline__ v8f wmma_bf16(v16bf a, v16bf b, v8f c) {
    return __builtin_amdgcn_wmma_f32_16x16x32_bf16(false, a, false, b, (short)0, c, false, false);
}

// ---------------- tiny elementwise kernels ----------------
__global__ void cvt_bf16_kernel(const float* __restrict__ src, bf16_t* __restrict__ dst, int n) {
    int i = blockIdx.x * 256 + threadIdx.x;
    if (i < n) dst[i] = f2bf(src[i]);
}

// one block per token row: out[row][0..255] = bf16(emb[token[row]][:])
__global__ void gather_embed_kernel(const int* __restrict__ tok, const float* __restrict__ emb,
                                    bf16_t* __restrict__ out) {
    int row = blockIdx.x;
    int d   = threadIdx.x;
    int t   = tok[row];
    out[(size_t)row * DD + d] = f2bf(emb[(size_t)t * DD + d]);
}

// ---------------- input-projection GEMM: gi = X(N x 256) * W^T + bias ----------------
// W stored bf16 row-major [768][256]; B-operand fragment rows = output column n.
// Block = 128x128 tile, 8 waves, each wave 32x64 (2x4 WMMA tiles), K=256.
__global__ __launch_bounds__(256) void proj_gemm_kernel(const bf16_t* __restrict__ X,
                                                        const bf16_t* __restrict__ W,
                                                        const float* __restrict__ bias,
                                                        float* __restrict__ out) {
    const int m0 = blockIdx.x * 128;
    const int n0 = blockIdx.y * 128;
    const int wave = threadIdx.x >> 5;
    const int wm = wave >> 1;          // 0..3 -> 32-row group
    const int wn = wave & 1;           // 0..1 -> 64-col group

    v8f acc[2][4];
#pragma unroll
    for (int i = 0; i < 2; ++i)
#pragma unroll
        for (int j = 0; j < 4; ++j) acc[i][j] = (v8f){0,0,0,0,0,0,0,0};

#pragma unroll 2
    for (int kc = 0; kc < DD / 32; ++kc) {
        v16bf a0 = load_frag16(X + (size_t)(m0 + wm * 32 + 0)  * DD + kc * 32, DD);
        v16bf a1 = load_frag16(X + (size_t)(m0 + wm * 32 + 16) * DD + kc * 32, DD);
#pragma unroll
        for (int j = 0; j < 4; ++j) {
            v16bf bb = load_frag16(W + (size_t)(n0 + wn * 64 + j * 16) * DD + kc * 32, DD);
            acc[0][j] = wmma_bf16(a0, bb, acc[0][j]);
            acc[1][j] = wmma_bf16(a1, bb, acc[1][j]);
        }
    }
    const int lane = threadIdx.x & 31;
    const int nn = lane & 15;
    const int mo = (lane >> 4) * 8;
#pragma unroll
    for (int i = 0; i < 2; ++i)
#pragma unroll
        for (int j = 0; j < 4; ++j) {
            int gn = n0 + wn * 64 + j * 16 + nn;
            float bia = bias[gn];
            int gm = m0 + wm * 32 + i * 16 + mo;
#pragma unroll
            for (int r = 0; r < 8; ++r)
                out[(size_t)(gm + r) * TD3 + gn] = acc[i][j][r] + bia;
        }
}

// ---------------- GRU recurrent scan ----------------
// One workgroup per (sequence, direction). 8 waves. Per step:
//   gh(32x768) = hB(32x256 bf16,in LDS) @ Whh^T  via WMMA, then gates in fp32.
// LDS: gh fp32 (96KB) + h fp32 (32KB) + h bf16 (16KB) = 144KB (<320KB/WGP).
__global__ __launch_bounds__(256) void gru_scan_kernel(
    const float* gi_sf, const float* gi_sb, const float* gi_tf, const float* gi_tb,
    const float* gi_cf, const float* gi_cb,
    const bf16_t* whhF, const bf16_t* whhB,
    const float* bhhF, const float* bhhB,
    bf16_t* e_src, bf16_t* e_tgt, bf16_t* e_cmnt,
    const float* w2) {
    extern __shared__ char smem[];
    float*  sGH = (float*)smem;               // [32][768]
    float*  sHF = sGH + 32 * TD3;             // [32][256]
    bf16_t* sHB = (bf16_t*)(sHF + 32 * DD);   // [32][256]

    const float* gi; const bf16_t* whh; const float* bhh;
    bf16_t* eout; const float* scale; int L; int rev;
    switch (blockIdx.x) {
        case 0: gi = gi_sf; whh = whhF; bhh = bhhF; eout = e_src;        scale = w2;        L = TT; rev = 0; break;
        case 1: gi = gi_sb; whh = whhB; bhh = bhhB; eout = e_src + DD;   scale = w2 + DD;   L = TT; rev = 1; break;
        case 2: gi = gi_tf; whh = whhF; bhh = bhhF; eout = e_tgt;        scale = w2;        L = TT; rev = 0; break;
        case 3: gi = gi_tb; whh = whhB; bhh = bhhB; eout = e_tgt + DD;   scale = w2 + DD;   L = TT; rev = 1; break;
        case 4: gi = gi_cf; whh = whhF; bhh = bhhF; eout = e_cmnt;       scale = nullptr;   L = JJ; rev = 0; break;
        default:gi = gi_cb; whh = whhB; bhh = bhhB; eout = e_cmnt + DD;  scale = nullptr;   L = JJ; rev = 1; break;
    }

    const int tid  = threadIdx.x;
    const int wave = tid >> 5;
    const int lane = tid & 31;
    const int nn = lane & 15;
    const int mo = (lane >> 4) * 8;

    for (int i = tid; i < 32 * DD; i += 256) { sHF[i] = 0.0f; sHB[i] = f2bf(0.0f); }
    __syncthreads();

    for (int s = 0; s < L; ++s) {
        const int t = rev ? (L - 1 - s) : s;

        // ---- gh = hB @ Whh^T : 2 M-tiles x 6 N-tiles per wave (cols wave*96..+95)
        v8f acc[2][6];
#pragma unroll
        for (int i = 0; i < 2; ++i)
#pragma unroll
            for (int j = 0; j < 6; ++j) acc[i][j] = (v8f){0,0,0,0,0,0,0,0};

#pragma unroll 2
        for (int kc = 0; kc < DD / 32; ++kc) {
            v16bf a0 = load_frag16(sHB + 0  * DD + kc * 32, DD);
            v16bf a1 = load_frag16(sHB + 16 * DD + kc * 32, DD);
#pragma unroll
            for (int j = 0; j < 6; ++j) {
                int nbase = wave * 96 + j * 16;
                v16bf bb = load_frag16(whh + (size_t)nbase * DD + kc * 32, DD);
                acc[0][j] = wmma_bf16(a0, bb, acc[0][j]);
                acc[1][j] = wmma_bf16(a1, bb, acc[1][j]);
            }
        }
#pragma unroll
        for (int i = 0; i < 2; ++i)
#pragma unroll
            for (int j = 0; j < 6; ++j) {
                int col = wave * 96 + j * 16 + nn;
                float bb = bhh[col];
#pragma unroll
                for (int r = 0; r < 8; ++r)
                    sGH[(i * 16 + mo + r) * TD3 + col] = acc[i][j][r] + bb;
            }
        __syncthreads();

        // ---- gates: thread tid handles dim d=tid for every batch b
        const int d = tid;
        for (int b = 0; b < BB; ++b) {
            const float* girow = gi + ((size_t)b * L + t) * TD3;
            float i_r = girow[d], i_z = girow[DD + d], i_n = girow[2 * DD + d];
            float h_r = sGH[b * TD3 + d];
            float h_z = sGH[b * TD3 + DD + d];
            float h_n = sGH[b * TD3 + 2 * DD + d];
            float r = sigm(i_r + h_r);
            float z = sigm(i_z + h_z);
            float n = tanhf(i_n + r * h_n);
            float h = sHF[b * DD + d];
            float hn = (1.0f - z) * n + z * h;
            sHF[b * DD + d] = hn;
            sHB[b * DD + d] = f2bf(hn);
            float sc = scale ? scale[d] : 1.0f;
            eout[((size_t)b * L + t) * TWO_D + d] = f2bf(hn * sc);
        }
        __syncthreads();
    }
}

// ---------------- similarity GEMM: S[b,t,j] = sum_d ectx[b,t,d]*ecm[b,j,d] + action*w2[512]
__global__ __launch_bounds__(256) void sim_gemm_kernel(const bf16_t* __restrict__ ectx,
                                                       const bf16_t* __restrict__ ecm,
                                                       const int* __restrict__ action,
                                                       const float* __restrict__ w2,
                                                       float* __restrict__ S) {
    const int b  = blockIdx.y;
    const int m0 = blockIdx.x * 128;
    const int wave = threadIdx.x >> 5;
    const int wm = wave >> 1;
    const int wn = wave & 1;
    const float w2l = w2[TWO_D];

    const bf16_t* Ab = ectx + ((size_t)b * TT + m0) * TWO_D;
    const bf16_t* Bb = ecm  + (size_t)b * JJ * TWO_D;

    v8f acc[2][4];
#pragma unroll
    for (int i = 0; i < 2; ++i)
#pragma unroll
        for (int j = 0; j < 4; ++j) acc[i][j] = (v8f){0,0,0,0,0,0,0,0};

    for (int kc = 0; kc < TWO_D / 32; ++kc) {
        v16bf a0 = load_frag16(Ab + (size_t)(wm * 32 + 0)  * TWO_D + kc * 32, TWO_D);
        v16bf a1 = load_frag16(Ab + (size_t)(wm * 32 + 16) * TWO_D + kc * 32, TWO_D);
#pragma unroll
        for (int j = 0; j < 4; ++j) {
            v16bf bb = load_frag16(Bb + (size_t)(wn * 64 + j * 16) * TWO_D + kc * 32, TWO_D);
            acc[0][j] = wmma_bf16(a0, bb, acc[0][j]);
            acc[1][j] = wmma_bf16(a1, bb, acc[1][j]);
        }
    }
    const int lane = threadIdx.x & 31;
    const int nn = lane & 15;
    const int mo = (lane >> 4) * 8;
#pragma unroll
    for (int i = 0; i < 2; ++i)
#pragma unroll
        for (int j = 0; j < 4; ++j) {
            int jj = wn * 64 + j * 16 + nn;
#pragma unroll
            for (int r = 0; r < 8; ++r) {
                int t = m0 + wm * 32 + i * 16 + mo + r;
                float add = (float)action[b * TT + t] * w2l;
                S[((size_t)b * TT + t) * JJ + jj] = acc[i][j][r] + add;
            }
        }
}

// ---------------- softmax over T of rowmax, then weighted sum over T ----------------
__global__ __launch_bounds__(256) void sim_reduce_kernel(const float* __restrict__ Ssrc,
                                                         const float* __restrict__ Stgt,
                                                         float* __restrict__ dout) {
    const int b   = blockIdx.x;
    const int dir = blockIdx.y;
    const float* S = dir ? Stgt : Ssrc;
    __shared__ float wrow[TT];
    __shared__ float red[256];
    const int tid = threadIdx.x;

    for (int t = tid; t < TT; t += 256) {
        const float* row = S + ((size_t)b * TT + t) * JJ;
        float m = -3.4e38f;
        for (int j = 0; j < JJ; ++j) m = fmaxf(m, row[j]);
        wrow[t] = m;
    }
    __syncthreads();
    float m = -3.4e38f;
    for (int t = tid; t < TT; t += 256) m = fmaxf(m, wrow[t]);
    red[tid] = m; __syncthreads();
    for (int off = 128; off > 0; off >>= 1) { if (tid < off) red[tid] = fmaxf(red[tid], red[tid + off]); __syncthreads(); }
    float M = red[0]; __syncthreads();
    float ssum = 0.0f;
    for (int t = tid; t < TT; t += 256) { float e = __expf(wrow[t] - M); wrow[t] = e; ssum += e; }
    red[tid] = ssum; __syncthreads();
    for (int off = 128; off > 0; off >>= 1) { if (tid < off) red[tid] += red[tid + off]; __syncthreads(); }
    float Z = red[0]; __syncthreads();

    const int j = tid & 127, hf = tid >> 7;
    float acc = 0.0f;
    for (int t = hf * 256; t < hf * 256 + 256; ++t)
        acc += wrow[t] * S[((size_t)b * TT + t) * JJ + j];
    red[tid] = acc; __syncthreads();
    if (hf == 0) {
        float v = (red[tid] + red[tid + 128]) / Z;
        dout[BB + b * 2 * JJ + dir * JJ + j] = v;   // S_diff at offset 32
    }
}

// ---------------- rank: result[b] = S_diff[b,:] . rank_w + rank_b ----------------
__global__ __launch_bounds__(256) void rank_kernel(float* __restrict__ dout,
                                                   const float* __restrict__ rw,
                                                   const float* __restrict__ rb) {
    __shared__ float red[256];
    const int tid = threadIdx.x;
    const int b = tid >> 3, l = tid & 7;
    const float* sd = dout + BB + (size_t)b * 2 * JJ;
    float a = 0.0f;
    for (int k = l; k < 2 * JJ; k += 8) a += sd[k] * rw[k];
    red[tid] = a; __syncthreads();
    if (l == 0) {
        float s = 0.0f;
        for (int q = 0; q < 8; ++q) s += red[b * 8 + q];
        dout[b] = s + rb[0];
    }
}

// ---------------- host launch ----------------
extern "C" void kernel_launch(void* const* d_in, const int* in_sizes, int n_in,
                              void* d_out, int out_size, void* d_ws, size_t ws_size,
                              hipStream_t stream) {
    (void)in_sizes; (void)n_in; (void)out_size; (void)ws_size;
    const int*   cmnt      = (const int*)d_in[0];
    const int*   src_token = (const int*)d_in[1];
    const int*   tgt_token = (const int*)d_in[2];
    const int*   src_act   = (const int*)d_in[3];
    const int*   tgt_act   = (const int*)d_in[4];
    const float* emb       = (const float*)d_in[5];
    const float* wih_f     = (const float*)d_in[6];
    const float* whh_f     = (const float*)d_in[7];
    const float* bih_f     = (const float*)d_in[8];
    const float* bhh_f     = (const float*)d_in[9];
    const float* wih_b     = (const float*)d_in[10];
    const float* whh_b     = (const float*)d_in[11];
    const float* bih_b     = (const float*)d_in[12];
    const float* bhh_b     = (const float*)d_in[13];
    const float* w2        = (const float*)d_in[14];
    const float* rank_w    = (const float*)d_in[15];
    const float* rank_b    = (const float*)d_in[16];
    float* out = (float*)d_out;

    // ---- workspace carve-out
    const size_t NW  = (size_t)TD3 * DD;          // 196608 per weight
    const size_t NST = (size_t)BB * TT;           // 16384 rows
    const size_t NC  = (size_t)BB * JJ;           // 4096 rows
    bf16_t* wihF = (bf16_t*)d_ws;
    bf16_t* wihB = wihF + NW;
    bf16_t* whhF = wihB + NW;
    bf16_t* whhB = whhF + NW;
    bf16_t* xsrc = whhB + NW;                     // NST*256
    bf16_t* xtgt = xsrc + NST * DD;
    bf16_t* xcm  = xtgt + NST * DD;               // NC*256
    bf16_t* esrc = xcm  + NC * DD;                // NST*512 (w2-scaled)
    bf16_t* etgt = esrc + NST * TWO_D;
    bf16_t* ecm  = etgt + NST * TWO_D;            // NC*512 (unscaled)
    float* fb    = (float*)(ecm + NC * TWO_D);
    float* gisf = fb;                             // NST*768 each
    float* gisb = gisf + NST * TD3;
    float* gitf = gisb + NST * TD3;
    float* gitb = gitf + NST * TD3;
    float* gicf = gitb + NST * TD3;               // NC*768 each
    float* gicb = gicf + NC * TD3;
    float* Ssrc = gicb + NC * TD3;                // B*T*J each
    float* Stgt = Ssrc + (size_t)BB * TT * JJ;

    // 1) weights -> bf16 (row-major [768][256], K contiguous)
    cvt_bf16_kernel<<<768, 256, 0, stream>>>(wih_f, wihF, (int)NW);
    cvt_bf16_kernel<<<768, 256, 0, stream>>>(wih_b, wihB, (int)NW);
    cvt_bf16_kernel<<<768, 256, 0, stream>>>(whh_f, whhF, (int)NW);
    cvt_bf16_kernel<<<768, 256, 0, stream>>>(whh_b, whhB, (int)NW);

    // 2) embedding gathers -> bf16
    gather_embed_kernel<<<(int)NST, 256, 0, stream>>>(src_token, emb, xsrc);
    gather_embed_kernel<<<(int)NST, 256, 0, stream>>>(tgt_token, emb, xtgt);
    gather_embed_kernel<<<(int)NC,  256, 0, stream>>>(cmnt,      emb, xcm);

    // 3) hoisted input projections gi = x@Wih^T + bih (big WMMA GEMMs)
    proj_gemm_kernel<<<dim3(128, 6), 256, 0, stream>>>(xsrc, wihF, bih_f, gisf);
    proj_gemm_kernel<<<dim3(128, 6), 256, 0, stream>>>(xsrc, wihB, bih_b, gisb);
    proj_gemm_kernel<<<dim3(128, 6), 256, 0, stream>>>(xtgt, wihF, bih_f, gitf);
    proj_gemm_kernel<<<dim3(128, 6), 256, 0, stream>>>(xtgt, wihB, bih_b, gitb);
    proj_gemm_kernel<<<dim3(32, 6),  256, 0, stream>>>(xcm,  wihF, bih_f, gicf);
    proj_gemm_kernel<<<dim3(32, 6),  256, 0, stream>>>(xcm,  wihB, bih_b, gicb);

    // 4) six concurrent recurrent scans (src/tgt/cmnt x fwd/bwd), LDS-resident h
    const int scanLds = 32 * TD3 * 4 + 32 * DD * 4 + 32 * DD * 2;  // 147456 B
    hipFuncSetAttribute((const void*)gru_scan_kernel,
                        hipFuncAttributeMaxDynamicSharedMemorySize, scanLds);
    gru_scan_kernel<<<6, 256, scanLds, stream>>>(gisf, gisb, gitf, gitb, gicf, gicb,
                                                 whhF, whhB, bhh_f, bhh_b,
                                                 esrc, etgt, ecm, w2);

    // 5) similarity GEMMs (w2 scaling already folded into esrc/etgt bf16)
    sim_gemm_kernel<<<dim3(4, 32), 256, 0, stream>>>(esrc, ecm, src_act, w2, Ssrc);
    sim_gemm_kernel<<<dim3(4, 32), 256, 0, stream>>>(etgt, ecm, tgt_act, w2, Stgt);

    // 6) rowmax -> softmax over T -> weighted sum; writes S_diff into d_out[32..]
    sim_reduce_kernel<<<dim3(32, 2), 256, 0, stream>>>(Ssrc, Stgt, out);

    // 7) rank GEMV; writes result into d_out[0..31]
    rank_kernel<<<1, 256, 0, stream>>>(out, rank_w, rank_b);
}
